// Window_Attention_59674275611077
// MI455X (gfx1250) — compile-verified, matrix operand order checked by reading, not run on hardware
//
#include <hip/hip_runtime.h>
#include <hip/hip_bf16.h>

// Window attention, CDNA5 (gfx1250), wave32 + v_wmma_f32_16x16x32_bf16
// + global_load_async_to_lds_b128 staging (ASYNCcnt) + global_prefetch_b8.
//
// Reference quirk: the (3C,B,L)->(B',nh,L,hd) reshape means for unit (b',h):
//   c    = b' / 8
//   base = (b'%8)*25088 + h*1568           (token offset)
//   Q' = Y[base .. base+1568, c]      as 49x32 (row-major l,d)
//   K' = Y[base .. , 512+c], V' = Y[base .., 1024+c]
// where Y = x @ w_qkv + b_qkv  (200704 x 1536, token-major).
// We store Y TRANSPOSED (Yt[c][t], bf16) so each Q'/K'/V' tile is contiguous.

#define MTOK 200704   // BNW * L = 4096*49
#define CDIM 512
#define N_QKV 1536
#define LWIN 49

typedef __bf16 bf16;
typedef __attribute__((ext_vector_type(16))) __bf16 v16bf;
typedef __attribute__((ext_vector_type(8)))  float  v8f;

// 16-bit A/B fragment K-index pattern (ISA 7.12.2): lane holds row/col (lane&15),
// half = lane>>4; VGPR v holds K = (v/4)*16 + half*8 + (v%4)*2 and +1.
__device__ __forceinline__ int frag_k0(int v, int half) {
    return ((v >> 2) << 4) | (half << 3) | ((v & 3) << 1);
}

// Load a fragment whose per-lane data is a contiguous 32-element bf16 row.
__device__ __forceinline__ v16bf frag_load_row(const bf16* rowptr, int kbase,
                                               int half, bool valid) {
    v16bf f;
#pragma unroll
    for (int v = 0; v < 8; ++v) {
        int k0 = kbase + frag_k0(v, half);
        f[2 * v]     = valid ? rowptr[k0]     : (bf16)0.0f;
        f[2 * v + 1] = valid ? rowptr[k0 + 1] : (bf16)0.0f;
    }
    return f;
}

__device__ __forceinline__ v8f wmma_bf16(v16bf a, v16bf b, v8f c) {
    return __builtin_amdgcn_wmma_f32_16x16x32_bf16(
        false, a, false, b, (short)0, c, false, false);
}

// Async global->LDS 16B transfer (per-lane). LDS address = low 32 bits of the
// generic shared pointer (flat LDS aperture keeps the offset in addr[31:0]).
__device__ __forceinline__ void async_copy_b128(void* lds_dst, const void* gsrc) {
    unsigned lds_addr = (unsigned)(unsigned long long)lds_dst;
    unsigned long long ga = (unsigned long long)gsrc;
    asm volatile("global_load_async_to_lds_b128 %0, %1, off"
                 :: "v"(lds_addr), "v"(ga) : "memory");
}
__device__ __forceinline__ void wait_async0() {
    asm volatile("s_wait_asynccnt 0x0" ::: "memory");
}

// ---------------------------------------------------------------------------
// Kernel 1: Yt[c][t] = (x @ w_qkv + b_qkv)^T, bf16 output.
// Block tile 64(M) x 256(N), K step 32; 8 waves (2M x 4N), wave tile 32x64
// = 2x4 WMMA tiles (8 WMMA per K-step per wave).
// ---------------------------------------------------------------------------
__global__ __launch_bounds__(256) void qkv_gemm_kernel(
    const float* __restrict__ x, const float* __restrict__ w,
    const float* __restrict__ bvec, bf16* __restrict__ yt)
{
    __shared__ __align__(16) bf16 As[64][32];    // [m][k]
    __shared__ __align__(16) bf16 Bs[256][32];   // [n][k] (K-pairs contiguous)

    const int tid  = threadIdx.x;
    const int wave = tid >> 5, lane = tid & 31;
    const int half = lane >> 4, l16 = lane & 15;
    const int waveM = wave >> 2, waveN = wave & 3;
    const int t0 = blockIdx.x * 64;
    const int n0 = blockIdx.y * 256;

    v8f acc[2][4] = {};

    const int am = tid >> 2, ak = (tid & 3) * 8;        // A: 64 x 32, 8/thread
    const int bk2 = (tid >> 4) * 2, bn16 = (tid & 15) * 16;  // B: K-pair x 16 n

    for (int kb = 0; kb < CDIM; kb += 32) {
        if (kb + 32 < CDIM) {
            __builtin_prefetch(x + (size_t)(t0 + am) * CDIM + (kb + 32) + ak, 0, 3);
            __builtin_prefetch(w + (size_t)(kb + 32 + bk2) * N_QKV + n0 + bn16, 0, 3);
        }
        // A: fp32 -> bf16, one b128 LDS store per thread
        const float4* apt =
            reinterpret_cast<const float4*>(x + (size_t)(t0 + am) * CDIM + kb + ak);
        float4 a0 = apt[0], a1 = apt[1];
        float av[8] = {a0.x, a0.y, a0.z, a0.w, a1.x, a1.y, a1.z, a1.w};
        union { bf16 h[8]; uint4 u4; } pa;
#pragma unroll
        for (int i = 0; i < 8; ++i) pa.h[i] = (bf16)av[i];
        *reinterpret_cast<uint4*>(&As[am][ak]) = pa.u4;

        // B: two adjacent k rows x 16 n -> packed b32 stores at Bs[n][k2]
        const float4* b0p =
            reinterpret_cast<const float4*>(w + (size_t)(kb + bk2) * N_QKV + n0 + bn16);
        const float4* b1p =
            reinterpret_cast<const float4*>(w + (size_t)(kb + bk2 + 1) * N_QKV + n0 + bn16);
        float4 q0 = b0p[0], q1 = b0p[1], q2 = b0p[2], q3 = b0p[3];
        float4 s0 = b1p[0], s1 = b1p[1], s2 = b1p[2], s3 = b1p[3];
        float r0[16] = {q0.x, q0.y, q0.z, q0.w, q1.x, q1.y, q1.z, q1.w,
                        q2.x, q2.y, q2.z, q2.w, q3.x, q3.y, q3.z, q3.w};
        float r1[16] = {s0.x, s0.y, s0.z, s0.w, s1.x, s1.y, s1.z, s1.w,
                        s2.x, s2.y, s2.z, s2.w, s3.x, s3.y, s3.z, s3.w};
#pragma unroll
        for (int i = 0; i < 16; ++i) {
            union { bf16 h[2]; unsigned u; } pb;
            pb.h[0] = (bf16)r0[i]; pb.h[1] = (bf16)r1[i];
            *reinterpret_cast<unsigned*>(&Bs[bn16 + i][bk2]) = pb.u;
        }
        __syncthreads();

        v16bf af[2], bfr[4];
#pragma unroll
        for (int i = 0; i < 2; ++i)
            af[i] = frag_load_row(&As[waveM * 32 + i * 16 + l16][0], 0, half, true);
#pragma unroll
        for (int j = 0; j < 4; ++j)
            bfr[j] = frag_load_row(&Bs[waveN * 64 + j * 16 + l16][0], 0, half, true);
#pragma unroll
        for (int i = 0; i < 2; ++i)
#pragma unroll
            for (int j = 0; j < 4; ++j)
                acc[i][j] = wmma_bf16(af[i], bfr[j], acc[i][j]);

        __syncthreads();
    }

    // Transposed epilogue: lane holds column c, 8 consecutive rows -> one b128.
#pragma unroll
    for (int i = 0; i < 2; ++i)
#pragma unroll
        for (int j = 0; j < 4; ++j) {
            int c   = n0 + waveN * 64 + j * 16 + l16;
            int m0g = t0 + waveM * 32 + i * 16 + 8 * half;
            float bias = bvec[c];
            union { bf16 h[8]; uint4 u4; } pk;
#pragma unroll
            for (int e = 0; e < 8; ++e) pk.h[e] = (bf16)(acc[i][j][e] + bias);
            *reinterpret_cast<uint4*>(yt + (size_t)c * MTOK + m0g) = pk.u4;
        }
}

// ---------------------------------------------------------------------------
// Kernel 2: fused attention per (b',h). Q/K/V tiles async-staged into LDS.
// 4 waves; wave w owns S/O rows 16w..16w+15.
// ---------------------------------------------------------------------------
__global__ __launch_bounds__(128) void attn_kernel(
    const bf16* __restrict__ yt, const float* __restrict__ mask,
    const float* __restrict__ biast, const int* __restrict__ idxt,
    bf16* __restrict__ z)
{
    __shared__ float bm[64][64];                 // bias + mask (pad = -1e30)
    __shared__ bf16  P[64][64];                  // softmax output (bf16)
    __shared__ __align__(16) bf16 Qs[1600];      // 49x32 staged (+pad)
    __shared__ __align__(16) bf16 Ks[1600];
    __shared__ __align__(16) bf16 Vs[1600];

    const int tid  = threadIdx.x;
    const int bp   = blockIdx.x >> 4;   // b'
    const int h    = blockIdx.x & 15;
    const int wmsk = bp & 63;

    const int cq = bp >> 3;
    const size_t base = (size_t)(bp & 7) * 25088 + (size_t)h * 1568;
    const bf16* Qp = yt + (size_t)cq * MTOK + base;
    const bf16* Kp = yt + (size_t)(CDIM + cq) * MTOK + base;
    const bf16* Vp = yt + (size_t)(2 * CDIM + cq) * MTOK + base;

    // Async-stage Q', K', V' (3 x 49x32 bf16 = 3 x 196 16-byte granules).
    for (int i = tid; i < 3 * 196; i += 128) {
        int mat = i / 196, idx = i - mat * 196;
        const bf16* g = (mat == 0 ? Qp : (mat == 1 ? Kp : Vp)) + idx * 8;
        bf16* l = (mat == 0 ? Qs : (mat == 1 ? Ks : Vs)) + idx * 8;
        async_copy_b128(l, g);
    }

    // bias + mask table (overlaps with async transfers)
    for (int i = tid; i < 64 * 64; i += 128) {
        int l = i >> 6, n = i & 63;
        float v = -1.0e30f;
        if (l < LWIN && n < LWIN)
            v = biast[idxt[l * LWIN + n] * 16 + h] +
                mask[(size_t)wmsk * (LWIN * LWIN) + l * LWIN + n];
        bm[l][n] = v;
    }

    wait_async0();
    __syncthreads();

    const int wave = tid >> 5, lane = tid & 31;
    const int half = lane >> 4, l16 = lane & 15;
    const int m0   = wave * 16;
    const int mrow = m0 + l16;

    // S = Q' K'^T : K = hd = 32 in a single WMMA per N-tile.
    v16bf aq = frag_load_row(Qs + mrow * 32, 0, half, mrow < LWIN);
    v8f s[4] = {};
#pragma unroll
    for (int j = 0; j < 4; ++j) {
        int n = j * 16 + l16;
        v16bf kf = frag_load_row(Ks + n * 32, 0, half, n < LWIN);
        s[j] = wmma_bf16(aq, kf, s[j]);
    }

    // Softmax: row m = 16 lanes (half-group) x 4 accumulators.
#pragma unroll
    for (int e = 0; e < 8; ++e) {
        int m = m0 + e + 8 * half;
        float v[4];
        float mx = -3.0e38f;
#pragma unroll
        for (int j = 0; j < 4; ++j) {
            v[j] = s[j][e] + bm[m][j * 16 + l16];
            mx = fmaxf(mx, v[j]);
        }
#pragma unroll
        for (int sft = 8; sft >= 1; sft >>= 1) mx = fmaxf(mx, __shfl_xor(mx, sft));
        float sum = 0.0f;
#pragma unroll
        for (int j = 0; j < 4; ++j) { v[j] = __expf(v[j] - mx); sum += v[j]; }
#pragma unroll
        for (int sft = 8; sft >= 1; sft >>= 1) sum += __shfl_xor(sum, sft);
        float inv = 1.0f / sum;
#pragma unroll
        for (int j = 0; j < 4; ++j)
            P[m][j * 16 + l16] = (bf16)(v[j] * inv);
    }
    // P rows are wave-private (written and read by the same wave; per-wave DS
    // ordering is in-order), so no block barrier is required here.

    // O = P @ V' : M=16, N=32 (2 tiles), K=64 (2 WMMA steps).
    v16bf ap0 = frag_load_row(&P[m0 + l16][0], 0,  half, true);
    v16bf ap1 = frag_load_row(&P[m0 + l16][0], 32, half, true);
#pragma unroll
    for (int j = 0; j < 2; ++j) {
        int d = j * 16 + l16;
        v16bf bv0, bv1;
#pragma unroll
        for (int v = 0; v < 8; ++v) {
            int k0 = frag_k0(v, half);
            bv0[2 * v]     = Vs[k0 * 32 + d];          // k < 32 always valid
            bv0[2 * v + 1] = Vs[(k0 + 1) * 32 + d];
            int k1 = 32 + k0;
            bv1[2 * v]     = (k1 < LWIN)     ? Vs[k1 * 32 + d]       : (bf16)0.0f;
            bv1[2 * v + 1] = (k1 + 1 < LWIN) ? Vs[(k1 + 1) * 32 + d] : (bf16)0.0f;
        }
        v8f o = {};
        o = wmma_bf16(ap0, bv0, o);
        o = wmma_bf16(ap1, bv1, o);
#pragma unroll
        for (int e = 0; e < 8; ++e) {
            int m = m0 + e + 8 * half;
            if (m < LWIN)
                z[((size_t)bp * LWIN + m) * CDIM + h * 32 + d] = (bf16)o[e];
        }
    }
}

// ---------------------------------------------------------------------------
// Kernel 3: out = Z @ w_proj + b_proj (fp32 out). A (bf16) async-staged.
// Block tile 64 x 256, wave tile 32x64.
// ---------------------------------------------------------------------------
__global__ __launch_bounds__(256) void proj_gemm_kernel(
    const bf16* __restrict__ zin, const float* __restrict__ w,
    const float* __restrict__ bvec, float* __restrict__ out)
{
    __shared__ __align__(16) bf16 As[64][32];
    __shared__ __align__(16) bf16 Bs[256][32];

    const int tid  = threadIdx.x;
    const int wave = tid >> 5, lane = tid & 31;
    const int half = lane >> 4, l16 = lane & 15;
    const int waveM = wave >> 2, waveN = wave & 3;
    const int t0 = blockIdx.x * 64;
    const int n0 = blockIdx.y * 256;

    v8f acc[2][4] = {};

    const int am = tid >> 2, ak = (tid & 3) * 8;
    const int bk2 = (tid >> 4) * 2, bn16 = (tid & 15) * 16;

    for (int kb = 0; kb < CDIM; kb += 32) {
        if (kb + 32 < CDIM)
            __builtin_prefetch(w + (size_t)(kb + 32 + bk2) * CDIM + n0 + bn16, 0, 3);

        // A: already bf16 -> async b128 straight into LDS (ASYNCcnt path)
        async_copy_b128(&As[am][ak], zin + (size_t)(t0 + am) * CDIM + kb + ak);

        // B: fp32 -> bf16 packed pair stores
        const float4* b0p =
            reinterpret_cast<const float4*>(w + (size_t)(kb + bk2) * CDIM + n0 + bn16);
        const float4* b1p =
            reinterpret_cast<const float4*>(w + (size_t)(kb + bk2 + 1) * CDIM + n0 + bn16);
        float4 q0 = b0p[0], q1 = b0p[1], q2 = b0p[2], q3 = b0p[3];
        float4 s0 = b1p[0], s1 = b1p[1], s2 = b1p[2], s3 = b1p[3];
        float r0[16] = {q0.x, q0.y, q0.z, q0.w, q1.x, q1.y, q1.z, q1.w,
                        q2.x, q2.y, q2.z, q2.w, q3.x, q3.y, q3.z, q3.w};
        float r1[16] = {s0.x, s0.y, s0.z, s0.w, s1.x, s1.y, s1.z, s1.w,
                        s2.x, s2.y, s2.z, s2.w, s3.x, s3.y, s3.z, s3.w};
#pragma unroll
        for (int i = 0; i < 16; ++i) {
            union { bf16 h[2]; unsigned u; } pb;
            pb.h[0] = (bf16)r0[i]; pb.h[1] = (bf16)r1[i];
            *reinterpret_cast<unsigned*>(&Bs[bn16 + i][bk2]) = pb.u;
        }

        wait_async0();
        __syncthreads();

        v16bf af[2], bfr[4];
#pragma unroll
        for (int i = 0; i < 2; ++i)
            af[i] = frag_load_row(&As[waveM * 32 + i * 16 + l16][0], 0, half, true);
#pragma unroll
        for (int j = 0; j < 4; ++j)
            bfr[j] = frag_load_row(&Bs[waveN * 64 + j * 16 + l16][0], 0, half, true);
#pragma unroll
        for (int i = 0; i < 2; ++i)
#pragma unroll
            for (int j = 0; j < 4; ++j)
                acc[i][j] = wmma_bf16(af[i], bfr[j], acc[i][j]);

        __syncthreads();
    }

#pragma unroll
    for (int i = 0; i < 2; ++i)
#pragma unroll
        for (int j = 0; j < 4; ++j) {
            int n   = n0 + waveN * 64 + j * 16 + l16;
            int m0g = t0 + waveM * 32 + i * 16 + 8 * half;
            float bias = bvec[n];
#pragma unroll
            for (int e = 0; e < 8; ++e)
                out[(size_t)(m0g + e) * CDIM + n] = acc[i][j][e] + bias;
        }
}

// ---------------------------------------------------------------------------
extern "C" void kernel_launch(void* const* d_in, const int* in_sizes, int n_in,
                              void* d_out, int out_size, void* d_ws, size_t ws_size,
                              hipStream_t stream) {
    const float* x          = (const float*)d_in[0];
    const float* mask       = (const float*)d_in[1];
    const float* w_qkv      = (const float*)d_in[2];
    const float* b_qkv      = (const float*)d_in[3];
    const float* w_proj     = (const float*)d_in[4];
    const float* b_proj     = (const float*)d_in[5];
    const float* bias_table = (const float*)d_in[6];
    const int*   index_tbl  = (const int*)d_in[7];
    float* out = (float*)d_out;

    // Workspace: Yt (1536 x 200704 bf16, ~617 MB) then Z (200704 x 512 bf16, ~206 MB)
    bf16* Yt = reinterpret_cast<bf16*>(d_ws);
    bf16* Z  = Yt + (size_t)N_QKV * MTOK;

    qkv_gemm_kernel<<<dim3(MTOK / 64, N_QKV / 256), 256, 0, stream>>>(
        x, w_qkv, b_qkv, Yt);
    attn_kernel<<<dim3(4096 * 16), 128, 0, stream>>>(
        Yt, mask, bias_table, index_tbl, Z);
    proj_gemm_kernel<<<dim3(MTOK / 64, CDIM / 256), 256, 0, stream>>>(
        Z, w_proj, b_proj, out);
}
